// MultiHeadAttn_25658134626329
// MI455X (gfx1250) — matrix operand order repaired and checked by reading, not compiled, not run on hardware
//
#include <hip/hip_runtime.h>
#include <hip/hip_bf16.h>
#include <stdint.h>

// Problem constants (from reference)
#define B_   4
#define T_   2048
#define D_   1024
#define H_   16
#define HD_  64

typedef __attribute__((ext_vector_type(16))) __bf16 bf16x16;
typedef __attribute__((ext_vector_type(8)))  float  f32x8;

union FragU {
    bf16x16 v;
    uint4   q[2];
};

__device__ __forceinline__ unsigned short f2bf(float f) {
    union { float f; unsigned int u; } x;
    x.f = f;
    unsigned int r = x.u + 0x7FFFu + ((x.u >> 16) & 1u);  // RNE
    return (unsigned short)(r >> 16);
}

// ---- WMMA fragment loaders (CDNA5 bf16 16x16x32 layouts, wave32) ----
// A (16x32): lane<16 holds row M=lane, K = [0..7] and [16..23] (+k0)
//            lane>=16 holds row M=lane-16, K = [8..15] and [24..31]
__device__ __forceinline__ bf16x16 load_fragA(const unsigned short* base, int ld,
                                              int m0, int k0, int lane) {
    int row = m0 + (lane & 15);
    int klo = k0 + ((lane & 16) ? 8 : 0);
    const unsigned short* p = base + row * ld + klo;
    FragU f;
    f.q[0] = *(const uint4*)(p);
    f.q[1] = *(const uint4*)(p + 16);
    return f.v;
}

// B (32x16): lane<16 holds col N=lane, K = [0..7] and [16..23] (+k0)
//            lane>=16 holds col N=lane-16, K = [8..15] and [24..31]
// Source is stored as Bt (N x K row-major), i.e. B[k][n] = Bt[n][k].
__device__ __forceinline__ bf16x16 load_fragB(const unsigned short* base, int ld,
                                              int n0, int k0, int lane) {
    int row = n0 + (lane & 15);
    int klo = k0 + ((lane & 16) ? 8 : 0);
    const unsigned short* p = base + row * ld + klo;
    FragU f;
    f.q[0] = *(const uint4*)(p);
    f.q[1] = *(const uint4*)(p + 16);
    return f.v;
}

__device__ __forceinline__ f32x8 wmma_bf16(bf16x16 a, bf16x16 b, f32x8 c) {
    return __builtin_amdgcn_wmma_f32_16x16x32_bf16(false, a, false, b,
                                                   (short)0, c, false, false);
}

__device__ __forceinline__ float hmax16(float x) {  // reduce within 16-lane half
    x = fmaxf(x, __shfl_xor(x, 1, 32));
    x = fmaxf(x, __shfl_xor(x, 2, 32));
    x = fmaxf(x, __shfl_xor(x, 4, 32));
    x = fmaxf(x, __shfl_xor(x, 8, 32));
    return x;
}
__device__ __forceinline__ float hsum16(float x) {
    x += __shfl_xor(x, 1, 32);
    x += __shfl_xor(x, 2, 32);
    x += __shfl_xor(x, 4, 32);
    x += __shfl_xor(x, 8, 32);
    return x;
}

// ---------------- f32 -> bf16 conversion ----------------
__global__ void cvt_f32_bf16(const float* __restrict__ src,
                             unsigned short* __restrict__ dst, int n) {
    int i = (blockIdx.x * blockDim.x + threadIdx.x) * 4;
    if (i + 3 < n) {
        float4 v = *(const float4*)(src + i);
        ushort4 o;
        o.x = f2bf(v.x); o.y = f2bf(v.y); o.z = f2bf(v.z); o.w = f2bf(v.w);
        *(ushort4*)(dst + i) = o;
    }
}

// ---------------- projection GEMM: out = A @ Wt^T + bias ----------------
// A:  M x K bf16 row-major, Wt: N x K bf16 row-major (torch weight layout)
// mode 0: write bf16, head layout  (B*H, T, HD)   [Q (scaled), K]
// mode 2: write bf16, transposed   (B*H, HD, T)   [V]
// mode 3: write f32,  row-major M x N             [final output]
__global__ __launch_bounds__(256) void proj_gemm(
    const unsigned short* __restrict__ A,
    const unsigned short* __restrict__ Wt,
    const float* __restrict__ bias,
    void* __restrict__ out,
    int M, int K, int N, int mode, float scale)
{
    const int lane = threadIdx.x & 31;
    const int wid  = threadIdx.x >> 5;
    const int wm   = wid & 3;
    const int wn   = wid >> 2;
    const int m0   = blockIdx.x * 256 + wm * 64;
    const int n0   = blockIdx.y * 128 + wn * 64;

    f32x8 acc[4][4];
#pragma unroll
    for (int i = 0; i < 4; ++i)
#pragma unroll
        for (int j = 0; j < 4; ++j)
#pragma unroll
            for (int r = 0; r < 8; ++r) acc[i][j][r] = 0.f;

    for (int k0 = 0; k0 < K; k0 += 32) {
        __builtin_prefetch(A  + (m0 + lane) * K + k0 + 32, 0, 1);
        __builtin_prefetch(Wt + (n0 + lane) * K + k0 + 32, 0, 1);
        bf16x16 a[4], b[4];
#pragma unroll
        for (int i = 0; i < 4; ++i) a[i] = load_fragA(A,  K, m0 + i * 16, k0, lane);
#pragma unroll
        for (int j = 0; j < 4; ++j) b[j] = load_fragB(Wt, K, n0 + j * 16, k0, lane);
#pragma unroll
        for (int i = 0; i < 4; ++i)
#pragma unroll
            for (int j = 0; j < 4; ++j)
                acc[i][j] = wmma_bf16(a[i], b[j], acc[i][j]);
    }

    const int half8 = (lane >> 4) << 3;   // 0 or 8
    const int ncl   = lane & 15;
#pragma unroll
    for (int j = 0; j < 4; ++j) {
        const int n  = n0 + j * 16 + ncl;
        const float bval = bias ? bias[n] : 0.f;
#pragma unroll
        for (int i = 0; i < 4; ++i) {
#pragma unroll
            for (int r = 0; r < 8; ++r) {
                const int m = m0 + i * 16 + r + half8;
                const float val = (acc[i][j][r] + bval) * scale;
                if (mode == 3) {
                    ((float*)out)[(size_t)m * N + n] = val;
                } else {
                    const int bb = m / T_, t = m % T_;
                    const int h = n >> 6, hd = n & 63;
                    size_t idx;
                    if (mode == 2)
                        idx = (((size_t)(bb * H_ + h)) * HD_ + hd) * T_ + t;
                    else
                        idx = (((size_t)(bb * H_ + h)) * T_ + t) * HD_ + hd;
                    ((unsigned short*)out)[idx] = f2bf(val);
                }
            }
        }
    }
}

// ---------------- flash attention over precomputed Qh/Kh/Vt ----------------
// Qh, Kh: (B*H, T, 64) bf16 (Q already scaled by 1/sqrt(DV))
// Vt:     (B*H, 64, T) bf16
// O:      (B*T, 1024) bf16 (heads re-concatenated)
__global__ __launch_bounds__(256) void attn_kernel(
    const unsigned short* __restrict__ Qh,
    const unsigned short* __restrict__ Kh,
    const unsigned short* __restrict__ Vt,
    const int* __restrict__ mask,
    unsigned short* __restrict__ O)
{
    __shared__ __align__(16) unsigned short plds[8][16 * 64];

    const int lane = threadIdx.x & 31;
    const int wid  = threadIdx.x >> 5;
    const int bh   = blockIdx.y;
    const int bb   = bh / H_;
    const int h    = bh % H_;
    const int q0   = blockIdx.x * 128 + wid * 16;   // query row base (within T)

    const unsigned short* Qb = Qh + (size_t)bh * T_ * HD_;
    const unsigned short* Kb = Kh + (size_t)bh * T_ * HD_;
    const unsigned short* Vb = Vt + (size_t)bh * HD_ * T_;
    const int* mrow = mask + bb * T_;

    bf16x16 aq0 = load_fragA(Qb, HD_, q0, 0,  lane);
    bf16x16 aq1 = load_fragA(Qb, HD_, q0, 32, lane);

    f32x8 acco[4];
    float mstat[8], lstat[8];
#pragma unroll
    for (int n = 0; n < 4; ++n)
#pragma unroll
        for (int r = 0; r < 8; ++r) acco[n][r] = 0.f;
#pragma unroll
    for (int r = 0; r < 8; ++r) { mstat[r] = -3.0e38f; lstat[r] = 0.f; }

    const int half8 = (lane >> 4) << 3;
    const int ncl   = lane & 15;

    for (int kt = 0; kt < T_; kt += 64) {
        // ---- scores S = Q K^T for a 16 x 64 tile ----
        f32x8 s[4];
#pragma unroll
        for (int j = 0; j < 4; ++j) {
            bf16x16 bk0 = load_fragB(Kb, HD_, kt + j * 16, 0,  lane);
            bf16x16 bk1 = load_fragB(Kb, HD_, kt + j * 16, 32, lane);
            f32x8 z;
#pragma unroll
            for (int r = 0; r < 8; ++r) z[r] = 0.f;
            s[j] = wmma_bf16(aq1, bk1, wmma_bf16(aq0, bk0, z));
            if (mrow[kt + j * 16 + ncl] == 0) {
#pragma unroll
                for (int r = 0; r < 8; ++r) s[j][r] = -65504.0f;
            }
        }
        // ---- online softmax (rows live across 16-lane halves) ----
        float alpha[8];
#pragma unroll
        for (int r = 0; r < 8; ++r) {
            float t = fmaxf(fmaxf(s[0][r], s[1][r]), fmaxf(s[2][r], s[3][r]));
            t = hmax16(t);
            float mn = fmaxf(mstat[r], t);
            alpha[r] = __expf(mstat[r] - mn);
            mstat[r] = mn;
        }
        float lsum[8];
#pragma unroll
        for (int r = 0; r < 8; ++r) lsum[r] = 0.f;
#pragma unroll
        for (int j = 0; j < 4; ++j)
#pragma unroll
            for (int r = 0; r < 8; ++r) {
                float p = __expf(s[j][r] - mstat[r]);
                s[j][r] = p;
                lsum[r] += p;
            }
#pragma unroll
        for (int r = 0; r < 8; ++r)
            lstat[r] = lstat[r] * alpha[r] + hsum16(lsum[r]);
#pragma unroll
        for (int n = 0; n < 4; ++n)
#pragma unroll
            for (int r = 0; r < 8; ++r) acco[n][r] *= alpha[r];

        // ---- restage P (C layout f32) -> A layout bf16 through LDS ----
#pragma unroll
        for (int j = 0; j < 4; ++j)
#pragma unroll
            for (int r = 0; r < 8; ++r)
                plds[wid][(r + half8) * 64 + j * 16 + ncl] = f2bf(s[j][r]);
        __syncthreads();
        bf16x16 ap0 = load_fragA((const unsigned short*)plds[wid], 64, 0, 0,  lane);
        bf16x16 ap1 = load_fragA((const unsigned short*)plds[wid], 64, 0, 32, lane);

        // ---- O += P V  (V transposed: B[k][n] = Vt[n][kt+k]) ----
#pragma unroll
        for (int n = 0; n < 4; ++n) {
            bf16x16 bv0 = load_fragB(Vb, T_, n * 16, kt,      lane);
            bf16x16 bv1 = load_fragB(Vb, T_, n * 16, kt + 32, lane);
            acco[n] = wmma_bf16(ap1, bv1, wmma_bf16(ap0, bv0, acco[n]));
        }
    }

    // ---- epilogue: normalize and store heads-concatenated ----
#pragma unroll
    for (int n = 0; n < 4; ++n) {
        const int col = h * 64 + n * 16 + ncl;
#pragma unroll
        for (int r = 0; r < 8; ++r) {
            const int t = q0 + r + half8;
            O[(size_t)(bb * T_ + t) * D_ + col] = f2bf(acco[n][r] / lstat[r]);
        }
    }
}

extern "C" void kernel_launch(void* const* d_in, const int* in_sizes, int n_in,
                              void* d_out, int out_size, void* d_ws, size_t ws_size,
                              hipStream_t stream) {
    const float* q    = (const float*)d_in[0];
    const float* k    = (const float*)d_in[1];
    const float* v    = (const float*)d_in[2];
    const int*   mask = (const int*)  d_in[3];
    const float* Wq   = (const float*)d_in[4];
    const float* bq   = (const float*)d_in[5];
    const float* Wk   = (const float*)d_in[6];
    const float* bk   = (const float*)d_in[7];
    const float* Wv   = (const float*)d_in[8];
    const float* bv   = (const float*)d_in[9];
    const float* Wo   = (const float*)d_in[10];
    const float* bo   = (const float*)d_in[11];
    float* out = (float*)d_out;

    const size_t SZ_X = (size_t)B_ * T_ * D_;   // 8388608 elements
    const size_t SZ_W = (size_t)D_ * D_;        // 1048576 elements

    unsigned short* ws  = (unsigned short*)d_ws;
    unsigned short* Xq  = ws;
    unsigned short* Xk  = Xq  + SZ_X;
    unsigned short* Xv  = Xk  + SZ_X;
    unsigned short* Wqb = Xv  + SZ_X;
    unsigned short* Wkb = Wqb + SZ_W;
    unsigned short* Wvb = Wkb + SZ_W;
    unsigned short* Wob = Wvb + SZ_W;
    unsigned short* Qh  = Wob + SZ_W;
    unsigned short* Kh  = Qh  + SZ_X;
    unsigned short* Vt  = Kh  + SZ_X;
    unsigned short* Ob  = Vt  + SZ_X;

    // 1) convert inputs + weights to bf16
    {
        int gx = (int)(SZ_X / (256 * 4));
        cvt_f32_bf16<<<gx, 256, 0, stream>>>(q, Xq, (int)SZ_X);
        cvt_f32_bf16<<<gx, 256, 0, stream>>>(k, Xk, (int)SZ_X);
        cvt_f32_bf16<<<gx, 256, 0, stream>>>(v, Xv, (int)SZ_X);
        int gw = (int)(SZ_W / (256 * 4));
        cvt_f32_bf16<<<gw, 256, 0, stream>>>(Wq, Wqb, (int)SZ_W);
        cvt_f32_bf16<<<gw, 256, 0, stream>>>(Wk, Wkb, (int)SZ_W);
        cvt_f32_bf16<<<gw, 256, 0, stream>>>(Wv, Wvb, (int)SZ_W);
        cvt_f32_bf16<<<gw, 256, 0, stream>>>(Wo, Wob, (int)SZ_W);
    }

    const int M = B_ * T_;   // 8192
    dim3 pgrid(M / 256, D_ / 128);

    // 2) projections (Q scaled by 1/sqrt(DV) = 1/32)
    proj_gemm<<<pgrid, 256, 0, stream>>>(Xq, Wqb, bq, Qh, M, D_, D_, 0, 1.0f / 32.0f);
    proj_gemm<<<pgrid, 256, 0, stream>>>(Xk, Wkb, bk, Kh, M, D_, D_, 0, 1.0f);
    proj_gemm<<<pgrid, 256, 0, stream>>>(Xv, Wvb, bv, Vt, M, D_, D_, 2, 1.0f);

    // 3) attention
    attn_kernel<<<dim3(T_ / 128, B_ * H_), 256, 0, stream>>>(Qh, Kh, Vt, mask, Ob);

    // 4) output projection -> f32 d_out
    proj_gemm<<<pgrid, 256, 0, stream>>>(Ob, Wob, bo, out, M, D_, D_, 3, 1.0f);
}
